// SyncQwen3VLMoeSparseMoeBlock_44418551775988
// MI455X (gfx1250) — matrix operand order repaired and checked by reading, not compiled, not run on hardware
//
#include <hip/hip_runtime.h>
#include <hip/hip_bf16.h>
#include <stdint.h>

// ---------------- problem constants (from reference) ----------------
#define TKN  8192      // B*S tokens
#define HD   2048      // hidden dim
#define ID   1408      // intermediate dim
#define NE   8         // experts
#define LDT  72        // LDS tile pitch in bf16 elems (64 data + 8 pad = 144B rows)

// ---------------- vector types ----------------
typedef float    f32x4 __attribute__((ext_vector_type(4)));
typedef unsigned u32x4 __attribute__((ext_vector_type(4)));
typedef int      i32x4 __attribute__((ext_vector_type(4)));
typedef int      i32x8 __attribute__((ext_vector_type(8)));
typedef __bf16   v16bf __attribute__((ext_vector_type(16)));
typedef __bf16   v8bf  __attribute__((ext_vector_type(8)));
typedef float    v8f   __attribute__((ext_vector_type(8)));

// ---------------- TDM availability probing ----------------
#if __has_include(<hip/amd_detail/amd_gfx1250_TDM.h>)
#define TDM_ARGS6 1
#else
#define TDM_ARGS6 0
#endif

#if defined(__HIP_DEVICE_COMPILE__) && \
    __has_builtin(__builtin_amdgcn_tensor_load_to_lds) && \
    __has_builtin(__builtin_amdgcn_s_wait_tensorcnt)
#define USE_TDM 1
#else
#define USE_TDM 0
#endif

#if USE_TDM
// 2D tile DMA: tile_dim0 = 64 bf16 (128B contiguous), tile_dim1 = 128 rows,
// LDS padding: after every 32 DWORDs (128B) insert 4 DWORDs (16B) -> 144B pitch.
// rows_valid < 128 -> TDM zero-fills OOB rows (tensor_dim1 bound).
__device__ __forceinline__ void tdm_tile_load(unsigned lds_byte_addr, const void* gptr,
                                              unsigned rows_valid, unsigned row_stride_elems) {
  unsigned long long ga = (unsigned long long)(uintptr_t)gptr;
  u32x4 g0 = { 1u,                                 // count=1, user mode
               lds_byte_addr,                      // lds_addr [63:32]
               (unsigned)ga,                       // global_addr lo
               (unsigned)(ga >> 32) | (2u << 30) };// global_addr hi | type=2
  unsigned td0 = 64u;                              // remaining elems, dim0 (always full)
  unsigned td1 = rows_valid;                       // remaining rows (zero-fill beyond)
  i32x8 g1 = {
    (int)((1u << 16) | (1u << 20) | (4u << 22) | (3u << 25)), // data_size=2B, pad_en, ivl=32DW, amt=4DW
    (int)((td0 & 0xffffu) << 16),                             // tensor_dim0 lo16 @ bit48
    (int)((td0 >> 16) | ((td1 & 0xffffu) << 16)),             // dim0 hi | dim1 lo
    (int)((td1 >> 16) | (64u << 16)),                         // dim1 hi | tile_dim0=64
    (int)128,                                                 // tile_dim1=128, tile_dim2=0
    (int)row_stride_elems,                                    // tensor_dim0_stride lo32
    0, 0
  };
  i32x4 z4 = {0,0,0,0};
#if TDM_ARGS6
  i32x8 z8 = {0,0,0,0,0,0,0,0};
  __builtin_amdgcn_tensor_load_to_lds(g0, g1, z4, z4, z8, 0);
#else
  __builtin_amdgcn_tensor_load_to_lds(g0, g1, z4, z4, 0);
#endif
}
#endif

__device__ __forceinline__ void tdm_wait() {
#if USE_TDM
  __builtin_amdgcn_s_wait_tensorcnt(0);
#endif
}

// ---------------- WMMA fragment loaders (from padded LDS tiles) ----------------
// A 16x32 bf16: lanes 0-15 = rows M, K chunks {0..7, 16..23}; lanes 16-31: {8..15, 24..31}
__device__ __forceinline__ v16bf ld_fragA(const __bf16* lds, int row0, int k0) {
  const int lane = threadIdx.x & 31;
  const __bf16* p = lds + (row0 + (lane & 15)) * LDT + k0 + ((lane >> 4) << 3);
  union { v16bf v; u32x4 q[2]; } f;
  f.q[0] = *(const u32x4*)(p);
  f.q[1] = *(const u32x4*)(p + 16);
  return f.v;
}
// B 32x16 bf16: lane = column N; lanes 0-15 hold K {0..7, 16..23}, lanes 16-31 hold K {8..15, 24..31}
__device__ __forceinline__ v16bf ld_fragB(const __bf16* lds, int col0, int k0) {
  const int lane = threadIdx.x & 31;
  const __bf16* p = lds + (col0 + (lane & 15)) * LDT + k0 + ((lane >> 4) << 3);
  union { v16bf v; u32x4 q[2]; } f;
  f.q[0] = *(const u32x4*)(p);
  f.q[1] = *(const u32x4*)(p + 16);
  return f.v;
}

// ---------------- kernels ----------------

// fp32 -> bf16 straight conversion (8 elems/thread)
__global__ void cvt_bf16_kernel(const float* __restrict__ src, __bf16* __restrict__ dst, long n) {
  long i = ((long)blockIdx.x * blockDim.x + threadIdx.x) * 8;
  if (i + 8 > n) return;
  f32x4 a = *(const f32x4*)(src + i);
  f32x4 b = *(const f32x4*)(src + i + 4);
  union { v8bf h; u32x4 q; } u;
  u.h[0]=(__bf16)a.x; u.h[1]=(__bf16)a.y; u.h[2]=(__bf16)a.z; u.h[3]=(__bf16)a.w;
  u.h[4]=(__bf16)b.x; u.h[5]=(__bf16)b.y; u.h[6]=(__bf16)b.z; u.h[7]=(__bf16)b.w;
  *(u32x4*)(dst + i) = u.q;
}

// fp32 [E][rows][cols] -> bf16 [E][cols][rows] tiled transpose+convert
__global__ __launch_bounds__(256) void transpose_cvt_kernel(const float* __restrict__ src,
                                                            __bf16* __restrict__ dst,
                                                            int rows, int cols) {
  __shared__ float tile[32][33];
  const int e = blockIdx.z;
  const size_t base = (size_t)e * rows * cols;
  const int c0 = blockIdx.x * 32, r0 = blockIdx.y * 32;
  const int tx = threadIdx.x, ty = threadIdx.y;
#pragma unroll
  for (int j = 0; j < 32; j += 8)
    tile[ty + j][tx] = src[base + (size_t)(r0 + ty + j) * cols + c0 + tx];
  __syncthreads();
#pragma unroll
  for (int j = 0; j < 32; j += 8)
    dst[base + (size_t)(c0 + ty + j) * rows + r0 + tx] = (__bf16)tile[tx][ty + j];
}

// routing: logits = x @ Wg, softmax, top-2, normalize; count tokens per expert
__global__ __launch_bounds__(256) void router_kernel(const float* __restrict__ x,
                                                     const float* __restrict__ Wg,
                                                     int* __restrict__ topi, float* __restrict__ topw,
                                                     unsigned* __restrict__ cnt) {
  __shared__ float sWg[HD * NE];  // 64 KB
  for (int i = threadIdx.x; i < HD * NE / 4; i += 256)
    ((f32x4*)sWg)[i] = ((const f32x4*)Wg)[i];
  __syncthreads();

  const int t = blockIdx.x * 256 + threadIdx.x;
  const float* xr = x + (size_t)t * HD;
  float acc[NE];
#pragma unroll
  for (int e = 0; e < NE; ++e) acc[e] = 0.f;
  for (int h = 0; h < HD; ++h) {
    const float xv = xr[h];
#pragma unroll
    for (int e = 0; e < NE; ++e) acc[e] = fmaf(xv, sWg[h * NE + e], acc[e]);
  }
  float m = acc[0];
#pragma unroll
  for (int e = 1; e < NE; ++e) m = fmaxf(m, acc[e]);
#pragma unroll
  for (int e = 0; e < NE; ++e) acc[e] = __expf(acc[e] - m);
  int i0 = 0; float v0 = acc[0];
#pragma unroll
  for (int e = 1; e < NE; ++e) if (acc[e] > v0) { v0 = acc[e]; i0 = e; }
  int i1 = (i0 == 0) ? 1 : 0; float v1 = acc[i1];
#pragma unroll
  for (int e = 0; e < NE; ++e)
    if (e != i0 && acc[e] > v1) { v1 = acc[e]; i1 = e; }
  const float inv = 1.f / (v0 + v1);   // softmax denom cancels in top-k renorm
  topi[2 * t] = i0;      topi[2 * t + 1] = i1;
  topw[2 * t] = v0 * inv; topw[2 * t + 1] = v1 * inv;
  atomicAdd(&cnt[i0], 1u);
  atomicAdd(&cnt[i1], 1u);
}

__global__ void offsets_kernel(const unsigned* __restrict__ cnt,
                               unsigned* __restrict__ offs, unsigned* __restrict__ cursor) {
  if (threadIdx.x < NE) cursor[threadIdx.x] = 0;
  if (threadIdx.x == 0) {
    unsigned run = 0;
    for (int e = 0; e < NE; ++e) { offs[e] = run; run += cnt[e]; }
  }
}

__global__ __launch_bounds__(256) void scatter_kernel(const int* __restrict__ topi,
                                                      const float* __restrict__ topw,
                                                      const unsigned* __restrict__ offs,
                                                      unsigned* __restrict__ cursor,
                                                      int* __restrict__ tok_idx,
                                                      float* __restrict__ tok_w) {
  const int t = blockIdx.x * 256 + threadIdx.x;
#pragma unroll
  for (int j = 0; j < 2; ++j) {
    const int e = topi[2 * t + j];
    const unsigned pos = atomicAdd(&cursor[e], 1u);
    const unsigned s = offs[e] + pos;
    tok_idx[s] = t;
    tok_w[s]   = topw[2 * t + j];
  }
}

// GEMM1: per-expert gathered x @ [Wgate | Wup] (bf16 WMMA, f32 acc) -> silu(g)*u -> act (bf16)
// weights pre-transposed to [E][I][H]. Double-buffered LDS, TDM prefetch of tile k+1
// overlapped with WMMA compute of tile k.
__global__ __launch_bounds__(256) void gemm1_kernel(const __bf16* __restrict__ xb,
                                                    const __bf16* __restrict__ wg_b,
                                                    const __bf16* __restrict__ wu_b,
                                                    const int* __restrict__ tok_idx,
                                                    const unsigned* __restrict__ offs,
                                                    const unsigned* __restrict__ cnt,
                                                    __bf16* __restrict__ act) {
  const int e = blockIdx.z;
  const unsigned count = cnt[e];
  const unsigned m0 = blockIdx.x * 128;
  if (m0 >= count) return;
  const unsigned base = offs[e] + m0;
  const int n0 = blockIdx.y * 128;

  __shared__ __bf16 sA [2][128 * LDT];
  __shared__ __bf16 sBg[2][128 * LDT];
  __shared__ __bf16 sBu[2][128 * LDT];
  __shared__ int sTok[128];

  const int tid  = threadIdx.x;
  const int wave = tid >> 5;
  const int wm   = wave >> 2;   // 0..1 -> 64-row slice
  const int wn   = wave & 3;    // 0..3 -> 32-col slice

  if (tid < 128) sTok[tid] = (m0 + (unsigned)tid < count) ? tok_idx[base + tid] : 0;
  __syncthreads();

  const v8f zf = {0.f,0.f,0.f,0.f,0.f,0.f,0.f,0.f};
  v8f accG[4][2], accU[4][2];
#pragma unroll
  for (int a = 0; a < 4; ++a)
#pragma unroll
    for (int b = 0; b < 2; ++b) { accG[a][b] = zf; accU[a][b] = zf; }

  const __bf16* gBgBase = wg_b + (size_t)e * ID * HD + (size_t)n0 * HD;
  const __bf16* gBuBase = wu_b + (size_t)e * ID * HD + (size_t)n0 * HD;

  auto stageA = [&](int buf, int kb) {
#pragma unroll
    for (int it = 0; it < 4; ++it) {
      const int q = it * 256 + tid;
      const int row = q >> 3, c8 = (q & 7) << 3;
      const __bf16* g = xb + (size_t)sTok[row] * HD + kb + c8;
      __builtin_prefetch(g + 64, 0, 0);
      *(u32x4*)(sA[buf] + row * LDT + c8) = *(const u32x4*)g;
    }
  };
  auto stageB = [&](int buf, int kb) {
#if USE_TDM
    if (wave == 0) tdm_tile_load((unsigned)(uintptr_t)(void*)sBg[buf], gBgBase + kb, 128, HD);
    if (wave == 1) tdm_tile_load((unsigned)(uintptr_t)(void*)sBu[buf], gBuBase + kb, 128, HD);
#else
#pragma unroll
    for (int it = 0; it < 4; ++it) {
      const int q = it * 256 + tid;
      const int row = q >> 3, c8 = (q & 7) << 3;
      *(u32x4*)(sBg[buf] + row * LDT + c8) = *(const u32x4*)(gBgBase + (size_t)row * HD + kb + c8);
      *(u32x4*)(sBu[buf] + row * LDT + c8) = *(const u32x4*)(gBuBase + (size_t)row * HD + kb + c8);
    }
#endif
  };

  // prologue: fill buffer 0
  stageA(0, 0);
  stageB(0, 0);
  tdm_wait();
  __syncthreads();

  const int NKB = HD / 64;
  for (int i = 0; i < NKB; ++i) {
    const int cur = i & 1, nxt = cur ^ 1;
    if (i + 1 < NKB) {        // prefetch next tile while computing current
      stageA(nxt, (i + 1) * 64);
      stageB(nxt, (i + 1) * 64);
    }

#pragma unroll
    for (int kk = 0; kk < 64; kk += 32) {
      v16bf af[4];
#pragma unroll
      for (int mf = 0; mf < 4; ++mf) af[mf] = ld_fragA(sA[cur], wm * 64 + mf * 16, kk);
#pragma unroll
      for (int nf = 0; nf < 2; ++nf) {
        const v16bf bg = ld_fragB(sBg[cur], wn * 32 + nf * 16, kk);
        const v16bf bu = ld_fragB(sBu[cur], wn * 32 + nf * 16, kk);
#pragma unroll
        for (int mf = 0; mf < 4; ++mf) {
          accG[mf][nf] = __builtin_amdgcn_wmma_f32_16x16x32_bf16(
              false, af[mf], false, bg, (short)0, accG[mf][nf], false, false);
          accU[mf][nf] = __builtin_amdgcn_wmma_f32_16x16x32_bf16(
              false, af[mf], false, bu, (short)0, accU[mf][nf], false, false);
        }
      }
    }
    tdm_wait();          // next tile's DMA complete
    __syncthreads();     // and everyone done reading 'cur'
  }

  // epilogue: SwiGLU, store bf16 activations per expert-slot
  const int lane = tid & 31;
  const int colBase = n0 + wn * 32 + (lane & 15);
  const int rowHalf = (lane >> 4) << 3;
#pragma unroll
  for (int mf = 0; mf < 4; ++mf)
#pragma unroll
    for (int nf = 0; nf < 2; ++nf)
#pragma unroll
      for (int j = 0; j < 8; ++j) {
        const int r = wm * 64 + mf * 16 + rowHalf + j;
        if (m0 + (unsigned)r < count) {
          const float g = accG[mf][nf][j];
          const float u = accU[mf][nf][j];
          const float a = (g / (1.f + __expf(-g))) * u;
          act[(size_t)(base + r) * ID + colBase + nf * 16] = (__bf16)a;
        }
      }
}

// GEMM2: act @ Wdown, scale by routing weight, atomic-add into out (fp32)
// wd_b pre-transposed to [E][H][I]. Double-buffered, TDM-prefetched pipeline.
__global__ __launch_bounds__(256) void gemm2_kernel(const __bf16* __restrict__ act,
                                                    const __bf16* __restrict__ wd_b,
                                                    const int* __restrict__ tok_idx,
                                                    const float* __restrict__ tok_w,
                                                    const unsigned* __restrict__ offs,
                                                    const unsigned* __restrict__ cnt,
                                                    float* __restrict__ out) {
  const int e = blockIdx.z;
  const unsigned count = cnt[e];
  const unsigned m0 = blockIdx.x * 128;
  if (m0 >= count) return;
  const unsigned base = offs[e] + m0;
  unsigned rv = count - m0; if (rv > 128u) rv = 128u;
  const int n0 = blockIdx.y * 128;

  __shared__ __bf16 sA[2][128 * LDT];
  __shared__ __bf16 sB[2][128 * LDT];
  __shared__ int   sTok[128];
  __shared__ float sW[128];

  const int tid  = threadIdx.x;
  const int wave = tid >> 5;
  const int wm   = wave >> 2;
  const int wn   = wave & 3;

  if (tid < 128) {
    const bool valid = m0 + (unsigned)tid < count;
    sTok[tid] = valid ? tok_idx[base + tid] : 0;
    sW[tid]   = valid ? tok_w[base + tid] : 0.f;
  }
  __syncthreads();

  const v8f zf = {0.f,0.f,0.f,0.f,0.f,0.f,0.f,0.f};
  v8f acc[4][2];
#pragma unroll
  for (int a = 0; a < 4; ++a)
#pragma unroll
    for (int b = 0; b < 2; ++b) acc[a][b] = zf;

  const __bf16* gABase = act + (size_t)base * ID;
  const __bf16* gBBase = wd_b + (size_t)e * HD * ID + (size_t)n0 * ID;

  auto stage = [&](int buf, int kb) {
#if USE_TDM
    if (wave == 0) tdm_tile_load((unsigned)(uintptr_t)(void*)sA[buf], gABase + kb, rv, ID);
    if (wave == 1) tdm_tile_load((unsigned)(uintptr_t)(void*)sB[buf], gBBase + kb, 128, ID);
#else
#pragma unroll
    for (int it = 0; it < 4; ++it) {
      const int q = it * 256 + tid;
      const int row = q >> 3, c8 = (q & 7) << 3;
      if ((unsigned)row < rv)
        *(u32x4*)(sA[buf] + row * LDT + c8) = *(const u32x4*)(gABase + (size_t)row * ID + kb + c8);
      else {
        const u32x4 z = {0u,0u,0u,0u};
        *(u32x4*)(sA[buf] + row * LDT + c8) = z;
      }
      *(u32x4*)(sB[buf] + row * LDT + c8) = *(const u32x4*)(gBBase + (size_t)row * ID + kb + c8);
    }
#endif
  };

  stage(0, 0);
  tdm_wait();
  __syncthreads();

  const int NKB = ID / 64;
  for (int i = 0; i < NKB; ++i) {
    const int cur = i & 1, nxt = cur ^ 1;
    if (i + 1 < NKB) stage(nxt, (i + 1) * 64);

#pragma unroll
    for (int kk = 0; kk < 64; kk += 32) {
      v16bf af[4];
#pragma unroll
      for (int mf = 0; mf < 4; ++mf) af[mf] = ld_fragA(sA[cur], wm * 64 + mf * 16, kk);
#pragma unroll
      for (int nf = 0; nf < 2; ++nf) {
        const v16bf bf = ld_fragB(sB[cur], wn * 32 + nf * 16, kk);
#pragma unroll
        for (int mf = 0; mf < 4; ++mf)
          acc[mf][nf] = __builtin_amdgcn_wmma_f32_16x16x32_bf16(
              false, af[mf], false, bf, (short)0, acc[mf][nf], false, false);
      }
    }
    tdm_wait();
    __syncthreads();
  }

  const int lane = tid & 31;
  const int colBase = n0 + wn * 32 + (lane & 15);
  const int rowHalf = (lane >> 4) << 3;
#pragma unroll
  for (int mf = 0; mf < 4; ++mf)
#pragma unroll
    for (int nf = 0; nf < 2; ++nf)
#pragma unroll
      for (int j = 0; j < 8; ++j) {
        const int r = wm * 64 + mf * 16 + rowHalf + j;
        if (m0 + (unsigned)r < count) {
          const float v = acc[mf][nf][j] * sW[r];
          atomicAdd(&out[(size_t)sTok[r] * HD + colBase + nf * 16], v);
        }
      }
}

// ---------------- host launcher ----------------
extern "C" void kernel_launch(void* const* d_in, const int* in_sizes, int n_in,
                              void* d_out, int out_size, void* d_ws, size_t ws_size,
                              hipStream_t stream) {
  (void)in_sizes; (void)n_in; (void)ws_size;
  const float* x     = (const float*)d_in[0];
  const float* Wg    = (const float*)d_in[1];
  const float* Wgate = (const float*)d_in[2];
  const float* Wup   = (const float*)d_in[3];
  const float* Wdown = (const float*)d_in[4];
  float* out = (float*)d_out;

  char* ws = (char*)d_ws;
  size_t off = 0;
  auto carve = [&](size_t bytes) -> void* {
    void* p = ws + off;
    off = (off + bytes + 255) & ~(size_t)255;
    return p;
  };
  __bf16* xb    = (__bf16*)carve((size_t)TKN * HD * 2);
  __bf16* wg_b  = (__bf16*)carve((size_t)NE * HD * ID * 2);
  __bf16* wu_b  = (__bf16*)carve((size_t)NE * HD * ID * 2);
  __bf16* wd_b  = (__bf16*)carve((size_t)NE * ID * HD * 2);
  __bf16* actb  = (__bf16*)carve((size_t)(2 * TKN + 128) * ID * 2);
  unsigned* cnt    = (unsigned*)carve(256);
  unsigned* offsb  = (unsigned*)carve(256);
  unsigned* cursor = (unsigned*)carve(256);
  int*   topi    = (int*)carve((size_t)TKN * 2 * 4);
  float* topw    = (float*)carve((size_t)TKN * 2 * 4);
  int*   tok_idx = (int*)carve((size_t)2 * TKN * 4);
  float* tok_w   = (float*)carve((size_t)2 * TKN * 4);

  hipMemsetAsync(out, 0, (size_t)out_size * sizeof(float), stream);
  hipMemsetAsync(cnt, 0, NE * sizeof(unsigned), stream);

  // bf16 conversion (+ N-major transpose of expert weights)
  cvt_bf16_kernel<<<(TKN * (size_t)HD) / (8 * 256), 256, 0, stream>>>(x, xb, (long)TKN * HD);
  transpose_cvt_kernel<<<dim3(ID / 32, HD / 32, NE), dim3(32, 8), 0, stream>>>(Wgate, wg_b, HD, ID);
  transpose_cvt_kernel<<<dim3(ID / 32, HD / 32, NE), dim3(32, 8), 0, stream>>>(Wup,   wu_b, HD, ID);
  transpose_cvt_kernel<<<dim3(HD / 32, ID / 32, NE), dim3(32, 8), 0, stream>>>(Wdown, wd_b, ID, HD);

  // routing + compaction
  router_kernel <<<TKN / 256, 256, 0, stream>>>(x, Wg, topi, topw, cnt);
  offsets_kernel<<<1, 32, 0, stream>>>(cnt, offsb, cursor);
  scatter_kernel<<<TKN / 256, 256, 0, stream>>>(topi, topw, offsb, cursor, tok_idx, tok_w);

  // expert GEMMs (worst-case grids; blocks early-exit past per-expert counts)
  gemm1_kernel<<<dim3(TKN / 128, ID / 128, NE), 256, 0, stream>>>(
      xb, wg_b, wu_b, tok_idx, offsb, cnt, actb);
  gemm2_kernel<<<dim3(TKN / 128, HD / 128, NE), 256, 0, stream>>>(
      actb, wd_b, tok_idx, tok_w, offsb, cnt, out);
}
